// LazyGGUFLinear_5093831213493
// MI455X (gfx1250) — matrix operand order repaired and checked by reading, not compiled, not run on hardware
//
#include <hip/hip_runtime.h>

// y[m,n] = sum_k x[m,k] * scales[n, k/32] * (codes[n,k] - 128)
// M = 2*2048 = 4096, N = 4096, K = 4096

#define K_DIM 4096
#define N_DIM 4096
#define NBLK  128

#define BM 128
#define BN 128
#define BK 64
#define LDT 72   // padded LDS row stride (bf16 elems): 144 bytes

typedef __attribute__((ext_vector_type(16))) __bf16 v16bf;
typedef __attribute__((ext_vector_type(8)))  __bf16 v8bf;
typedef __attribute__((ext_vector_type(8)))  float  v8f;
typedef __attribute__((ext_vector_type(4)))  float  f4;
typedef __attribute__((ext_vector_type(4)))  int    i4;

#define SHUF16(lo, hi) \
  __builtin_shufflevector((lo), (hi), 0,1,2,3,4,5,6,7,8,9,10,11,12,13,14,15)
#define SHUF8(lo, hi) __builtin_shufflevector((lo), (hi), 0,1,2,3,4,5,6,7)

// A fragment (16x32 bf16, MxK): lane half lk selects K chunks
// [kk + lk*8, +8) and [kk + 16 + lk*8, +8) of row base.
__device__ __forceinline__ v16bf ldfragA(const unsigned short* rowbase, int kk, int lk) {
  v8bf lo = *reinterpret_cast<const v8bf*>(rowbase + kk + lk * 8);
  v8bf hi = *reinterpret_cast<const v8bf*>(rowbase + kk + 16 + lk * 8);
  return SHUF16(lo, hi);
}

// B fragment (32x16 bf16, KxN): lane holds column n = lane%16,
// K chunks [kk + lk*8, +8) and [kk + 16 + lk*8, +8) (same order as A).
__device__ __forceinline__ v16bf ldfragB(const unsigned short* rowbase, int kk, int lk) {
  v8bf lo = *reinterpret_cast<const v8bf*>(rowbase + kk + lk * 8);
  v8bf hi = *reinterpret_cast<const v8bf*>(rowbase + kk + 16 + lk * 8);
  return SHUF16(lo, hi);
}

__global__ __launch_bounds__(256)
void gguf_q8_linear_wmma(const float* __restrict__ x,
                         const float* __restrict__ scales,
                         const int*   __restrict__ codes,
                         float* __restrict__ out) {
  // Double-buffered tiles (constant indices 0/1 -> immediate ds offsets).
  __shared__ alignas(16) unsigned short As[2][BM * LDT];
  __shared__ alignas(16) unsigned short Bs[2][BN * LDT];

  const int tid  = threadIdx.x;
  const int lane = tid & 31;
  const int wave = tid >> 5;
  const int wm = wave >> 2;        // 0..1  -> 64-row slice of C
  const int wn = wave & 3;         // 0..3  -> 32-col slice of C
  const int lm = lane & 15;
  const int lk = lane >> 4;

  const int blockM = blockIdx.y * BM;
  const int blockN = blockIdx.x * BN;

  // Cooperative tile loading: thread owns (row, 32-elem K half) of the tile.
  const int lrow = tid >> 1;       // 0..127
  const int lkh  = tid & 1;        // 0..1

  const float* xrow = x      + (size_t)(blockM + lrow) * K_DIM + lkh * 32;
  const int*   crow = codes  + (size_t)(blockN + lrow) * K_DIM + lkh * 32;
  const float* srow = scales + (size_t)(blockN + lrow) * NBLK  + lkh;

  v8f acc[4][2] = {};

  f4 areg[8];
  i4 breg[8];
  float bscale;

  auto fetchTile = [&](int kt) {
    const f4* pa = reinterpret_cast<const f4*>(xrow + kt);
    const i4* pb = reinterpret_cast<const i4*>(crow + kt);
#pragma unroll
    for (int i = 0; i < 8; ++i) { areg[i] = pa[i]; breg[i] = pb[i]; }
    bscale = srow[kt >> 5];
  };

  auto storeTile = [&](int b) {
    unsigned short* da = &As[b][lrow * LDT + lkh * 32];
    unsigned short* db = &Bs[b][lrow * LDT + lkh * 32];
    const float s  = bscale;
    const float ns = -128.0f * s;   // (c-128)*s == fma(c, s, -128s)
    const v8f sv  = {s, s, s, s, s, s, s, s};
    const v8f nsv = {ns, ns, ns, ns, ns, ns, ns, ns};
#pragma unroll
    for (int g = 0; g < 4; ++g) {
      // x: f32 -> bf16 (vector fptrunc; backend may use v_cvt_pk_bf16_f32)
      v8f av = SHUF8(areg[2 * g], areg[2 * g + 1]);
      v8bf ab = __builtin_convertvector(av, v8bf);
      *reinterpret_cast<v8bf*>(da + g * 8) = ab;

      // codes: i32 -> f32, fma dequant (candidate v_pk_fma_f32), -> bf16
      f4 f0 = __builtin_convertvector(breg[2 * g], f4);
      f4 f1 = __builtin_convertvector(breg[2 * g + 1], f4);
      v8f fv = SHUF8(f0, f1);
      fv = __builtin_elementwise_fma(fv, sv, nsv);
      v8bf bb = __builtin_convertvector(fv, v8bf);
      *reinterpret_cast<v8bf*>(db + g * 8) = bb;
    }
  };

  auto computeTile = [&](int b) {
#pragma unroll
    for (int kk = 0; kk < BK; kk += 32) {
      v16bf af[4], bfr[2];
#pragma unroll
      for (int mi = 0; mi < 4; ++mi)
        af[mi] = ldfragA(&As[b][(wm * 64 + mi * 16 + lm) * LDT], kk, lk);
#pragma unroll
      for (int ni = 0; ni < 2; ++ni)
        bfr[ni] = ldfragB(&Bs[b][(wn * 32 + ni * 16 + lm) * LDT], kk, lk);
#pragma unroll
      for (int mi = 0; mi < 4; ++mi)
#pragma unroll
        for (int ni = 0; ni < 2; ++ni)
          acc[mi][ni] = __builtin_amdgcn_wmma_f32_16x16x32_bf16(
              false, af[mi], false, bfr[ni], (short)0, acc[mi][ni],
              false, false);
    }
  };

  // Prologue: tile 0 -> regs -> buffer 0.
  fetchTile(0);
  storeTile(0);

  // Steady state: branch-free, two tiles per iteration, constant buffers.
  // Invariant at loop top: buf0 holds tile kt.
  for (int kt = 0; kt < K_DIM - 2 * BK; kt += 2 * BK) {
    fetchTile(kt + BK);
    if (kt + 3 * BK < K_DIM) {                  // warm L2 (global_prefetch_b8)
      __builtin_prefetch(xrow + kt + 3 * BK, 0, 1);
      __builtin_prefetch(crow + kt + 3 * BK, 0, 1);
    }
    __syncthreads();
    computeTile(0);
    storeTile(1);

    fetchTile(kt + 2 * BK);
    if (kt + 4 * BK < K_DIM) {
      __builtin_prefetch(xrow + kt + 4 * BK, 0, 1);
      __builtin_prefetch(crow + kt + 4 * BK, 0, 1);
    }
    __syncthreads();
    computeTile(1);
    storeTile(0);
  }

  // Epilogue: tiles K-2BK (buf0) and K-BK (buf1).
  fetchTile(K_DIM - BK);
  __syncthreads();
  computeTile(0);
  storeTile(1);
  __syncthreads();
  computeTile(1);

  // C/D 16x16 f32 layout: lanes 0-15 -> N=lm, M=r; lanes 16-31 -> M=r+8.
#pragma unroll
  for (int mi = 0; mi < 4; ++mi) {
#pragma unroll
    for (int ni = 0; ni < 2; ++ni) {
#pragma unroll
      for (int r = 0; r < 8; ++r) {
        int row = blockM + wm * 64 + mi * 16 + lk * 8 + r;
        int col = blockN + wn * 32 + ni * 16 + lm;
        out[(size_t)row * N_DIM + col] = acc[mi][ni][r];
      }
    }
  }
}

extern "C" void kernel_launch(void* const* d_in, const int* in_sizes, int n_in,
                              void* d_out, int out_size, void* d_ws, size_t ws_size,
                              hipStream_t stream) {
  const float* x      = (const float*)d_in[0];   // [2,2048,4096] f32
  const float* scales = (const float*)d_in[1];   // [4096,128]    f32
  const int*   codes  = (const int*)d_in[2];     // [4096,128,32] i32
  float* out = (float*)d_out;                    // [2,2048,4096] f32

  const int M = in_sizes[0] / K_DIM;             // 4096
  dim3 grid(N_DIM / BN, M / BM);                 // (32, 32)
  gguf_q8_linear_wmma<<<grid, 256, 0, stream>>>(x, scales, codes, out);
}
